// TrueVectorizedSlidingWindowModel_4002909520748
// MI455X (gfx1250) — matrix operand order repaired and checked by reading, not compiled, not run on hardware
//
#include <hip/hip_runtime.h>
#include <hip/hip_bf16.h>

typedef __bf16 bf16_t;
typedef __attribute__((ext_vector_type(16))) __bf16 v16bf;
typedef __attribute__((ext_vector_type(8)))  float  v8f;

#define SEQ 2048
#define DM  512
#define HD  64
#define NH  8
#define WND 128

// ---------- bf16 split helpers (manual RNE, bit-exact) ----------
__device__ inline bf16_t f2bf(float x) {
    unsigned u = __builtin_bit_cast(unsigned, x);
    unsigned r = (u + 0x7FFFu + ((u >> 16) & 1u)) >> 16;
    unsigned short s = (unsigned short)r;
    return __builtin_bit_cast(bf16_t, s);
}
__device__ inline float bf2f(bf16_t b) {
    unsigned short s = __builtin_bit_cast(unsigned short, b);
    unsigned u = ((unsigned)s) << 16;
    return __builtin_bit_cast(float, u);
}
__device__ inline void split2(float x, bf16_t& hi, bf16_t& lo) {
    bf16_t h = f2bf(x);
    hi = h;
    lo = f2bf(x - bf2f(h));
}

// ---------- WMMA operand loaders (contiguous k = 16h + j order, wave32) ----------
// A 16x32 bf16: lane L (m=L&15, h=L>>4): elems j = A[m][h*16 + j]
__device__ inline v16bf load_a_tile(const bf16_t* base, int ld, int lane) {
    int m = lane & 15, h = lane >> 4;
    const bf16_t* p0 = base + m * ld + h * 16;
    v16bf a;
#pragma unroll
    for (int j = 0; j < 16; ++j) { a[j] = p0[j]; }
    return a;
}
// B 32x16 bf16: lane L (n=L&15, h=L>>4): elems j = B[h*16 + j][n].
// Caller stores B transposed (Bt[n][k] row-major) so this is a contiguous 16-element run.
__device__ inline v16bf load_b_tile(const bf16_t* bt_base, int ld, int lane) {
    int n = lane & 15, h = lane >> 4;
    const bf16_t* p = bt_base + n * ld + h * 16;
    v16bf b;
#pragma unroll
    for (int j = 0; j < 16; ++j) b[j] = p[j];
    return b;
}

// bf16x3 fp32-accurate product-accumulate: C += Ah*Bh + Ah*Bl + Al*Bh
__device__ inline v8f wmma3(v16bf ah, v16bf al, v16bf bh, v16bf bl, v8f acc) {
    acc = __builtin_amdgcn_wmma_f32_16x16x32_bf16(false, ah, false, bh, (short)0, acc, false, false);
    acc = __builtin_amdgcn_wmma_f32_16x16x32_bf16(false, ah, false, bl, (short)0, acc, false, false);
    acc = __builtin_amdgcn_wmma_f32_16x16x32_bf16(false, al, false, bh, (short)0, acc, false, false);
    return acc;
}

// One wave computes a 32x64 tile of C = A[M][K] @ Bt[N][K]^T with K-chunks of 32.
// Two A tiles share every B tile -> ~33 FLOP per L2 byte in the hot loop.
__device__ inline void gemm_tile_32x64(const bf16_t* Ahi, const bf16_t* Alo, int lda,
                                       const bf16_t* Bthi, const bf16_t* Btlo, int ldb,
                                       int m0, int n0, int K, int lane, v8f acc[2][4]) {
#pragma unroll 1
    for (int k0 = 0; k0 < K; k0 += 32) {
        v16bf ah0 = load_a_tile(Ahi + (size_t)m0 * lda + k0, lda, lane);
        v16bf al0 = load_a_tile(Alo + (size_t)m0 * lda + k0, lda, lane);
        v16bf ah1 = load_a_tile(Ahi + (size_t)(m0 + 16) * lda + k0, lda, lane);
        v16bf al1 = load_a_tile(Alo + (size_t)(m0 + 16) * lda + k0, lda, lane);
#pragma unroll
        for (int u = 0; u < 4; ++u) {
            v16bf bh = load_b_tile(Bthi + (size_t)(n0 + 16 * u) * ldb + k0, ldb, lane);
            v16bf bl = load_b_tile(Btlo + (size_t)(n0 + 16 * u) * ldb + k0, ldb, lane);
            acc[0][u] = wmma3(ah0, al0, bh, bl, acc[0][u]);
            acc[1][u] = wmma3(ah1, al1, bh, bl, acc[1][u]);
        }
    }
}

// ---------- prep kernels ----------
__global__ void k_prep_x(const float* __restrict__ x, bf16_t* __restrict__ hi,
                         bf16_t* __restrict__ lo, int n) {
    int i = blockIdx.x * blockDim.x + threadIdx.x;
    if (i >= n) return;
    bf16_t h, l; split2(x[i], h, l);
    hi[i] = h; lo[i] = l;
}
// W is [K=512][N=512] row-major; write Wt[n][k] (transposed) hi/lo
__global__ void k_prep_w(const float* __restrict__ W, bf16_t* __restrict__ hi,
                         bf16_t* __restrict__ lo) {
    int i = blockIdx.x * blockDim.x + threadIdx.x;
    if (i >= DM * DM) return;
    int k = i >> 9, n = i & 511;
    bf16_t h, l; split2(W[i], h, l);
    hi[n * DM + k] = h; lo[n * DM + k] = l;
}

// ---------- QKV projection: Y = X @ [Wq|Wk|Wv]  (N=1536), epilogue splits per-head ----------
__global__ __launch_bounds__(256) void k_gemm_qkv(
    const bf16_t* __restrict__ Xhi, const bf16_t* __restrict__ Xlo,
    const bf16_t* __restrict__ Wthi, const bf16_t* __restrict__ Wtlo,
    bf16_t* __restrict__ Qhi, bf16_t* __restrict__ Qlo,
    bf16_t* __restrict__ Khi, bf16_t* __restrict__ Klo,
    bf16_t* __restrict__ Vthi, bf16_t* __restrict__ Vtlo) {
    int lane = threadIdx.x & 31;
    int wave = blockIdx.x * (blockDim.x >> 5) + (threadIdx.x >> 5);
    if (wave >= (SEQ / 32) * 24) return;
    int mtile = wave / 24, nt = wave % 24;
    int m0 = mtile * 32;
    v8f acc[2][4] = {};
    gemm_tile_32x64(Xhi, Xlo, DM, Wthi, Wtlo, DM, m0, nt * 64, DM, lane, acc);
    int t = nt >> 3, head = nt & 7;           // each 64-col tile is exactly one head
    int hh = lane >> 4, n = lane & 15;
#pragma unroll
    for (int mi = 0; mi < 2; ++mi)
#pragma unroll
        for (int u = 0; u < 4; ++u)
#pragma unroll
            for (int r = 0; r < 8; ++r) {
                float v = acc[mi][u][r];
                int s = m0 + mi * 16 + hh * 8 + r;
                int d = u * 16 + n;
                bf16_t hi, lo; split2(v, hi, lo);
                if (t == 0) { size_t o = ((size_t)head * SEQ + s) * HD + d; Qhi[o] = hi; Qlo[o] = lo; }
                else if (t == 1) { size_t o = ((size_t)head * SEQ + s) * HD + d; Khi[o] = hi; Klo[o] = lo; }
                else { size_t o = ((size_t)head * HD + d) * SEQ + s; Vthi[o] = hi; Vtlo[o] = lo; }
            }
}

// ---------- sliding-window attention: one wave per (head, 16-query block) ----------
__global__ __launch_bounds__(128) void k_attn(
    const bf16_t* __restrict__ Qhi, const bf16_t* __restrict__ Qlo,
    const bf16_t* __restrict__ Khi, const bf16_t* __restrict__ Klo,
    const bf16_t* __restrict__ Vthi, const bf16_t* __restrict__ Vtlo,
    bf16_t* __restrict__ Ahi_out, bf16_t* __restrict__ Alo_out) {
    __shared__ float lds[4][16][132];
    int lane = threadIdx.x & 31;
    int wv = threadIdx.x >> 5;
    int wave = blockIdx.x * 4 + wv;          // 1024 waves = 8 heads * 128 qblocks
    int head = wave >> 7, qb = wave & 127;
    int m0 = qb * 16;
    const bf16_t* Qh = Qhi + (size_t)head * SEQ * HD;
    const bf16_t* Ql = Qlo + (size_t)head * SEQ * HD;
    const bf16_t* Kh = Khi + (size_t)head * SEQ * HD;
    const bf16_t* Kl = Klo + (size_t)head * SEQ * HD;
    const bf16_t* Vh = Vthi + (size_t)head * HD * SEQ;
    const bf16_t* Vl = Vtlo + (size_t)head * HD * SEQ;

    // Q A-operands, K-dim 64 = 2 chunks of 32
    v16bf qh0 = load_a_tile(Qh + (size_t)m0 * HD + 0, HD, lane);
    v16bf qh1 = load_a_tile(Qh + (size_t)m0 * HD + 32, HD, lane);
    v16bf ql0 = load_a_tile(Ql + (size_t)m0 * HD + 0, HD, lane);
    v16bf ql1 = load_a_tile(Ql + (size_t)m0 * HD + 32, HD, lane);

    int kbase = m0 - 112;                    // 8 key tiles: m0-112 .. m0 (9th always fully masked)
    v8f sc[8];
#pragma unroll
    for (int t = 0; t < 8; ++t) {
        int kb = kbase + t * 16;
        int kbc = kb < 0 ? 0 : kb;           // clamped reads are masked below
        // K rows row-major are exactly the B-operand for Q @ K^T
        v16bf bh0 = load_b_tile(Kh + (size_t)kbc * HD + 0, HD, lane);
        v16bf bl0 = load_b_tile(Kl + (size_t)kbc * HD + 0, HD, lane);
        v16bf bh1 = load_b_tile(Kh + (size_t)kbc * HD + 32, HD, lane);
        v16bf bl1 = load_b_tile(Kl + (size_t)kbc * HD + 32, HD, lane);
        v8f a = {};
        a = wmma3(qh0, ql0, bh0, bl0, a);
        a = wmma3(qh1, ql1, bh1, bl1, a);
        sc[t] = a;
    }
    int hh = lane >> 4, n = lane & 15;
    const float scale = 0.125f;              // 1/sqrt(64)
    const float NEG = -3.0e38f;
#pragma unroll
    for (int t = 0; t < 8; ++t) {
        int kb = kbase + t * 16;
#pragma unroll
        for (int r = 0; r < 8; ++r) {
            int s = m0 + hh * 8 + r;
            int k = kb + n;
            bool ok = (k >= 0) && (k <= s) && (k >= s - (WND - 1));
            sc[t][r] = ok ? sc[t][r] * scale : NEG;
        }
    }
    // softmax: each C-layout row spans 16 lanes (fixed hh, fixed VGPR r)
    float rinv[8];
#pragma unroll
    for (int r = 0; r < 8; ++r) {
        float mx = sc[0][r];
#pragma unroll
        for (int t = 1; t < 8; ++t) mx = fmaxf(mx, sc[t][r]);
#pragma unroll
        for (int d = 1; d < 16; d <<= 1) mx = fmaxf(mx, __shfl_xor(mx, d, 32));
        float sum = 0.f;
#pragma unroll
        for (int t = 0; t < 8; ++t) { float e = __expf(sc[t][r] - mx); sc[t][r] = e; sum += e; }
#pragma unroll
        for (int d = 1; d < 16; d <<= 1) sum += __shfl_xor(sum, d, 32);
        rinv[r] = 1.0f / sum;
    }
    // bounce P (16x128) through LDS to reshape C-layout -> A-layout
#pragma unroll
    for (int t = 0; t < 8; ++t)
#pragma unroll
        for (int r = 0; r < 8; ++r)
            lds[wv][hh * 8 + r][t * 16 + n] = sc[t][r] * rinv[r];
    __syncthreads();

    v8f o[4] = {};
#pragma unroll
    for (int c = 0; c < 4; ++c) {            // key dim 128 = 4 chunks of 32
        v16bf pah, pal;
        {
            int m = lane & 15, h2 = lane >> 4;
#pragma unroll
            for (int j = 0; j < 16; ++j) {
                bf16_t hi, lo;
                split2(lds[wv][m][32 * c + h2 * 16 + j], hi, lo);
                pah[j] = hi; pal[j] = lo;
            }
        }
#pragma unroll
        for (int u = 0; u < 4; ++u) {        // output dim 64 = 4 N-tiles
            v16bf bh, bl;
            {
                int nn = lane & 15, h2 = lane >> 4;
                int kstart = kbase + 32 * c + h2 * 16;   // 16-aligned run; entire run <0 when OOB
                if (kstart < 0) kstart = 0;              // P is exactly 0 there
                const bf16_t* ph = Vh + (size_t)(u * 16 + nn) * SEQ + kstart;
                const bf16_t* pl = Vl + (size_t)(u * 16 + nn) * SEQ + kstart;
#pragma unroll
                for (int j = 0; j < 16; ++j) { bh[j] = ph[j]; bl[j] = pl[j]; }
            }
            o[u] = wmma3(pah, pal, bh, bl, o[u]);
        }
    }
#pragma unroll
    for (int u = 0; u < 4; ++u)
#pragma unroll
        for (int r = 0; r < 8; ++r) {
            int s = m0 + hh * 8 + r;
            int d = head * HD + u * 16 + n;
            bf16_t hi, lo; split2(o[u][r], hi, lo);
            Ahi_out[(size_t)s * DM + d] = hi;
            Alo_out[(size_t)s * DM + d] = lo;
        }
}

// ---------- output projection: d_out = attn @ Wo ----------
__global__ __launch_bounds__(256) void k_gemm_out(
    const bf16_t* __restrict__ Ahi, const bf16_t* __restrict__ Alo,
    const bf16_t* __restrict__ Wthi, const bf16_t* __restrict__ Wtlo,
    float* __restrict__ out) {
    int lane = threadIdx.x & 31;
    int wave = blockIdx.x * (blockDim.x >> 5) + (threadIdx.x >> 5);
    if (wave >= (SEQ / 32) * (DM / 64)) return;
    int mtile = wave / 8, nt = wave % 8;
    int m0 = mtile * 32;
    v8f acc[2][4] = {};
    gemm_tile_32x64(Ahi, Alo, DM, Wthi, Wtlo, DM, m0, nt * 64, DM, lane, acc);
    int hh = lane >> 4, n = lane & 15;
#pragma unroll
    for (int mi = 0; mi < 2; ++mi)
#pragma unroll
        for (int u = 0; u < 4; ++u)
#pragma unroll
            for (int r = 0; r < 8; ++r) {
                int s = m0 + mi * 16 + hh * 8 + r;
                int d = nt * 64 + u * 16 + n;
                out[(size_t)s * DM + d] = acc[mi][u][r];
            }
}

extern "C" void kernel_launch(void* const* d_in, const int* in_sizes, int n_in,
                              void* d_out, int out_size, void* d_ws, size_t ws_size,
                              hipStream_t stream) {
    const float* x  = (const float*)d_in[0];
    const float* Wq = (const float*)d_in[1];
    const float* Wk = (const float*)d_in[2];
    const float* Wv = (const float*)d_in[3];
    const float* Wo = (const float*)d_in[4];
    float* out = (float*)d_out;

    char* ws = (char*)d_ws;
    size_t off = 0;
    auto take = [&](size_t bytes) { char* p = ws + off; off += (bytes + 255) & ~(size_t)255; return p; };
    const size_t XB  = (size_t)SEQ * DM * sizeof(bf16_t);      // 2 MB
    const size_t WQB = (size_t)3 * DM * DM * sizeof(bf16_t);   // 1.5 MB
    const size_t WOB = (size_t)DM * DM * sizeof(bf16_t);
    const size_t HB  = (size_t)NH * SEQ * HD * sizeof(bf16_t); // 2 MB

    bf16_t* Xhi   = (bf16_t*)take(XB);
    bf16_t* Xlo   = (bf16_t*)take(XB);
    bf16_t* WtQhi = (bf16_t*)take(WQB);   // [1536][512]: rows 0-511 Wq^T, 512-1023 Wk^T, 1024-1535 Wv^T
    bf16_t* WtQlo = (bf16_t*)take(WQB);
    bf16_t* WtOhi = (bf16_t*)take(WOB);
    bf16_t* WtOlo = (bf16_t*)take(WOB);
    bf16_t* Qhi   = (bf16_t*)take(HB);
    bf16_t* Qlo   = (bf16_t*)take(HB);
    bf16_t* Khi   = (bf16_t*)take(HB);
    bf16_t* Klo   = (bf16_t*)take(HB);
    bf16_t* Vthi  = (bf16_t*)take(HB);
    bf16_t* Vtlo  = (bf16_t*)take(HB);
    bf16_t* AtHi  = (bf16_t*)take(XB);
    bf16_t* AtLo  = (bf16_t*)take(XB);
    (void)ws_size; (void)n_in; (void)in_sizes; (void)out_size;

    // 1) split x to bf16 hi/lo
    k_prep_x<<<(SEQ * DM + 255) / 256, 256, 0, stream>>>(x, Xhi, Xlo, SEQ * DM);
    // 2) transpose+split weights into B-operand layout
    k_prep_w<<<(DM * DM + 255) / 256, 256, 0, stream>>>(Wq, WtQhi + 0 * DM * DM, WtQlo + 0 * DM * DM);
    k_prep_w<<<(DM * DM + 255) / 256, 256, 0, stream>>>(Wk, WtQhi + 1 * DM * DM, WtQlo + 1 * DM * DM);
    k_prep_w<<<(DM * DM + 255) / 256, 256, 0, stream>>>(Wv, WtQhi + 2 * DM * DM, WtQlo + 2 * DM * DM);
    k_prep_w<<<(DM * DM + 255) / 256, 256, 0, stream>>>(Wo, WtOhi, WtOlo);
    // 3) fused QKV projection (1536 waves of 32x64 tiles, 8 waves/block)
    k_gemm_qkv<<<192, 256, 0, stream>>>(Xhi, Xlo, WtQhi, WtQlo, Qhi, Qlo, Khi, Klo, Vthi, Vtlo);
    // 4) sliding-window attention (1024 waves, 4 waves/block for LDS)
    k_attn<<<256, 128, 0, stream>>>(Qhi, Qlo, Khi, Klo, Vthi, Vtlo, AtHi, AtLo);
    // 5) output projection (512 waves)
    k_gemm_out<<<64, 256, 0, stream>>>(AtHi, AtLo, WtOhi, WtOlo, out);
}
